// TransformerLayer_17403207483584
// MI455X (gfx1250) — matrix operand order repaired and checked by reading, not compiled, not run on hardware
//
#include <hip/hip_runtime.h>

#define B_ 2
#define L_ 2048
#define E_ 1024
#define H_ 16
#define DH_ 64
#define R_ 128
#define FF_ 4096
#define M_ (B_*L_)
#define EPS_ 1e-5f

typedef __attribute__((ext_vector_type(16))) __bf16 v16bf;
typedef __attribute__((ext_vector_type(8)))  float  v8f;
typedef __attribute__((ext_vector_type(4)))  unsigned int v4u;
typedef __attribute__((ext_vector_type(8)))  int v8i;
typedef __attribute__((ext_vector_type(4)))  int v4i;

#if defined(__HIP_DEVICE_COMPILE__) && __has_builtin(__builtin_amdgcn_tensor_load_to_lds)
#define HAVE_TDM 1
#else
#define HAVE_TDM 0
#endif

union Frag16 { v16bf v; uint4 q[2]; unsigned short u[16]; };
union V64    { uint4 q[4]; unsigned short u[32]; };

__device__ __forceinline__ unsigned short f2bf(float f) {
  unsigned u = __float_as_uint(f);
  u += 0x7fffu + ((u >> 16) & 1u);
  return (unsigned short)(u >> 16);
}

__device__ __forceinline__ v8f vzero8() {
  v8f z;
#pragma unroll
  for (int i = 0; i < 8; i++) z[i] = 0.f;
  return z;
}

#if HAVE_TDM
// Issue a TDM 2D tile load: 128 rows x 32 bf16 elements, row stride K elements,
// into LDS with 4-DWORD padding after every 16 DWORDs (LDS row stride 40 halves).
// D# packing per CDNA5 ISA.
__device__ __forceinline__ void tdm_load_tile_128x32(unsigned lds_byte_off,
                                                     const unsigned short* gaddr,
                                                     unsigned K) {
  unsigned long long ga = (unsigned long long)(size_t)gaddr;
  v4u g0;
  g0[0] = 1u;                                           // count=1, user descriptor
  g0[1] = lds_byte_off;                                 // lds_addr (bytes)
  g0[2] = (unsigned)(ga & 0xffffffffu);                 // global_addr[31:0]
  g0[3] = (unsigned)((ga >> 32) & 0x01ffffffu) | (2u << 30);  // addr[56:32], type=2
  const unsigned td0 = K;                               // tensor_dim0 (elements)
  const unsigned td1 = 1u << 20;                        // tensor_dim1 (rows, generous)
  v8i g1;
  g1[0] = (int)((1u << 16)      /* data_size = 2 bytes            */
              | (1u << 20)      /* pad_enable                     */
              | (3u << 22)      /* pad_interval: 16 DWORDs        */
              | (3u << 25));    /* pad_amount: 4 DWORDs           */
  g1[1] = (int)((td0 & 0xffffu) << 16);                 // [63:48] = tensor_dim0 lo
  g1[2] = (int)(((td0 >> 16) & 0xffffu) | ((td1 & 0xffffu) << 16));
  g1[3] = (int)(((td1 >> 16) & 0xffffu) | (32u << 16)); // tile_dim0 = 32
  g1[4] = (int)128u;                                    // tile_dim1 = 128, tile_dim2 = 0
  g1[5] = (int)(K & 0xffffffffu);                       // tensor_dim0_stride lo
  g1[6] = 0;                                            // stride hi / dim1_stride lo
  g1[7] = 0;
  v4i gz = (v4i)0;
#if defined(__clang_major__) && (__clang_major__ >= 23)
  v8i gz8 = (v8i)0;
  __builtin_amdgcn_tensor_load_to_lds(g0, g1, gz, gz, gz8, 0);
#else
  __builtin_amdgcn_tensor_load_to_lds(g0, g1, gz, gz, 0);
#endif
}
#endif

// ---------------------------------------------------------------- f32 -> bf16
__global__ void k_cvt(const float* __restrict__ in, unsigned short* __restrict__ out, int n) {
  for (int i = blockIdx.x * blockDim.x + threadIdx.x; i < n; i += gridDim.x * blockDim.x)
    out[i] = f2bf(in[i]);
}

// ---------------------------------------------------------------- LayerNorm -> bf16
__global__ __launch_bounds__(256) void k_ln(const float* __restrict__ x,
                                            const float* __restrict__ g,
                                            const float* __restrict__ b,
                                            unsigned short* __restrict__ out) {
  const int row = blockIdx.x;
  const int tid = threadIdx.x;
  const float* xr = x + (size_t)row * E_;
  float s = 0.f, ss = 0.f;
  for (int c = tid; c < E_; c += 256) { float v = xr[c]; s += v; ss += v * v; }
#pragma unroll
  for (int off = 16; off; off >>= 1) { s += __shfl_xor(s, off, 32); ss += __shfl_xor(ss, off, 32); }
  __shared__ float rb[16];
  if ((tid & 31) == 0) { rb[tid >> 5] = s; rb[8 + (tid >> 5)] = ss; }
  __syncthreads();
  float ts = 0.f, tss = 0.f;
#pragma unroll
  for (int i = 0; i < 8; i++) { ts += rb[i]; tss += rb[8 + i]; }
  const float mean = ts * (1.f / E_);
  const float var  = tss * (1.f / E_) - mean * mean;
  const float rstd = rsqrtf(var + EPS_);
  unsigned short* orow = out + (size_t)row * E_;
  for (int c = tid; c < E_; c += 256)
    orow[c] = f2bf((xr[c] - mean) * rstd * g[c] + b[c]);
}

// ---------------------------------------------------------------- WMMA GEMM
// C[M,N] = A[M,K](bf16) * Bw[N,K]^T(bf16) + bias, optional relu / residual.
// Block tile 128x128, BK=32, TDM double-buffered LDS staging.
// 8 waves: each wave computes 32x64 (2x4 WMMA tiles).
template <bool RELU, bool RESID, bool OUTF, bool OUTB>
__global__ __launch_bounds__(256) void k_gemm(const unsigned short* __restrict__ A,
                                              const unsigned short* __restrict__ Bw,
                                              const float* __restrict__ bias,
                                              const float* __restrict__ resid,
                                              float* __restrict__ Cf,
                                              unsigned short* __restrict__ Cb,
                                              int M, int N, int K) {
  __shared__ __attribute__((aligned(16))) unsigned short lA[2][128 * 40];
  __shared__ __attribute__((aligned(16))) unsigned short lB[2][128 * 40];
  const int t = threadIdx.x;
  const int lane = t & 31, w = t >> 5;
  const int hi = lane >> 4, ln = lane & 15;
  const int wm = (w & 3) * 32;
  const int wn = (w >> 2) * 64;
  const int m0 = blockIdx.x * 128;
  const int n0 = blockIdx.y * 128;

  const unsigned short* Abase = A + (size_t)m0 * K;
  const unsigned short* Bbase = Bw + (size_t)n0 * K;

  v8f acc[2][4];
#pragma unroll
  for (int i = 0; i < 2; i++)
#pragma unroll
    for (int j = 0; j < 4; j++) acc[i][j] = vzero8();

#if HAVE_TDM
  if (w == 0) {  // prologue: stage 0 into buffer 0 via the Tensor Data Mover
    tdm_load_tile_128x32((unsigned)(size_t)&lA[0][0], Abase, K);
    tdm_load_tile_128x32((unsigned)(size_t)&lB[0][0], Bbase, K);
  }
#endif

  for (int kt = 0; kt < K; kt += 32) {
    const int buf = (kt >> 5) & 1;
    __syncthreads();  // all waves done reading the buffer TDM is about to overwrite
#if HAVE_TDM
    if (w == 0) {
      if (kt + 32 < K) {
        tdm_load_tile_128x32((unsigned)(size_t)&lA[buf ^ 1][0], Abase + kt + 32, K);
        tdm_load_tile_128x32((unsigned)(size_t)&lB[buf ^ 1][0], Bbase + kt + 32, K);
        __builtin_amdgcn_s_wait_tensorcnt(2);  // in-order: current stage resident
      } else {
        __builtin_amdgcn_s_wait_tensorcnt(0);
      }
    }
#else
    {  // fallback: per-thread staged copy
      const int lrow = t >> 1, lch = (t & 1) * 16;
      const uint4* pa = (const uint4*)(Abase + (size_t)lrow * K + kt + lch);
      const uint4* pb = (const uint4*)(Bbase + (size_t)lrow * K + kt + lch);
      uint4 a0 = pa[0], a1 = pa[1];
      uint4 b0 = pb[0], b1 = pb[1];
      *(uint4*)&lA[buf][lrow * 40 + lch]     = a0;
      *(uint4*)&lA[buf][lrow * 40 + lch + 8] = a1;
      *(uint4*)&lB[buf][lrow * 40 + lch]     = b0;
      *(uint4*)&lB[buf][lrow * 40 + lch + 8] = b1;
    }
#endif
    __syncthreads();  // publish staged tiles

    Frag16 af[2], bf[4];
#pragma unroll
    for (int i = 0; i < 2; i++) {  // A 16x32: lane row=ln, k = hi*8 + (e<8?e:e+8)
      const int row = wm + i * 16 + ln;
      af[i].q[0] = *(const uint4*)&lA[buf][row * 40 + hi * 8];
      af[i].q[1] = *(const uint4*)&lA[buf][row * 40 + hi * 8 + 16];
    }
#pragma unroll
    for (int j = 0; j < 4; j++) {  // B 32x16: lane col=ln, k = hi*8 + (e<8?e:e+8)
      const int col = wn + j * 16 + ln;
      bf[j].q[0] = *(const uint4*)&lB[buf][col * 40 + hi * 8];
      bf[j].q[1] = *(const uint4*)&lB[buf][col * 40 + hi * 8 + 16];
    }
#pragma unroll
    for (int i = 0; i < 2; i++)
#pragma unroll
      for (int j = 0; j < 4; j++)
        acc[i][j] = __builtin_amdgcn_wmma_f32_16x16x32_bf16(
            false, af[i].v, false, bf[j].v, (short)0, acc[i][j], false, false);
  }

  // epilogue: C layout row = r + 8*hi, col = ln
#pragma unroll
  for (int i = 0; i < 2; i++) {
#pragma unroll
    for (int j = 0; j < 4; j++) {
      const int n = n0 + wn + j * 16 + ln;
      const float bsv = bias[n];
#pragma unroll
      for (int r = 0; r < 8; r++) {
        const int m = m0 + wm + i * 16 + r + 8 * hi;
        float v = acc[i][j][r] + bsv;
        if (RELU)  v = fmaxf(v, 0.f);
        if (RESID) v += resid[(size_t)m * N + n];
        if (OUTF)  Cf[(size_t)m * N + n] = v;
        if (OUTB)  Cb[(size_t)m * N + n] = f2bf(v);
      }
    }
  }
}

// ---------------------------------------------------------------- banded attention
// One wave per (b, h, 16-query tile). Window R=128 => 9 key tiles of 16.
__global__ __launch_bounds__(32) void k_attn(const unsigned short* __restrict__ q,
                                             const unsigned short* __restrict__ kk,
                                             const unsigned short* __restrict__ v,
                                             const float* __restrict__ res_pos,
                                             unsigned short* __restrict__ o) {
  __shared__ __attribute__((aligned(16))) unsigned short Pl[16 * 16];
  __shared__ __attribute__((aligned(16))) unsigned short Vt[64 * 24];
  const int lane = threadIdx.x;
  const int hi = lane >> 4, ln = lane & 15;
  const int nt = L_ / 16;
  const int bid = blockIdx.x;
  const int it = bid % nt;
  const int h  = (bid / nt) % H_;
  const int b  = bid / (nt * H_);
  const int i0 = it * 16;

  // Q A-fragments: row = ln, dim k = f*32 + hi*8 + (e<8?e:e+8)
  Frag16 aq[2];
  {
    const unsigned short* qr = q + ((size_t)(b * L_ + i0 + ln) * E_ + h * DH_);
#pragma unroll
    for (int f = 0; f < 2; f++) {
      aq[f].q[0] = *(const uint4*)(qr + f * 32 + hi * 8);
      aq[f].q[1] = *(const uint4*)(qr + f * 32 + hi * 8 + 16);
    }
  }

  float rmax[8], rsum[8];
  v8f oacc[4];
#pragma unroll
  for (int r = 0; r < 8; r++) { rmax[r] = -1e30f; rsum[r] = 0.f; }
#pragma unroll
  for (int g2 = 0; g2 < 4; g2++) oacc[g2] = vzero8();

  for (int jt = 0; jt < 9; jt++) {
    const int j0 = i0 - 128 + jt * 16;
    const int jk = j0 + ln;
    const int jkc = jk < 0 ? 0 : (jk > L_ - 1 ? L_ - 1 : jk);

    // K B-fragments: col = ln (key), k-dim = f*32 + hi*8 + (e<8?e:e+8)
    Frag16 bk[2];
    {
      const unsigned short* kr = kk + ((size_t)(b * L_ + jkc) * E_ + h * DH_);
#pragma unroll
      for (int f = 0; f < 2; f++) {
        bk[f].q[0] = *(const uint4*)(kr + f * 32 + hi * 8);
        bk[f].q[1] = *(const uint4*)(kr + f * 32 + hi * 8 + 16);
      }
    }
    v8f s = vzero8();
    s = __builtin_amdgcn_wmma_f32_16x16x32_bf16(false, aq[0].v, false, bk[0].v, (short)0, s, false, false);
    s = __builtin_amdgcn_wmma_f32_16x16x32_bf16(false, aq[1].v, false, bk[1].v, (short)0, s, false, false);

    __syncthreads();  // protect previous Vt/Pl readers
    {  // stage V tile transposed: Vt[dim][key], key = ln, dims hi*32..+31
      const unsigned short* vr = v + ((size_t)(b * L_ + jkc) * E_ + h * DH_ + hi * 32);
      V64 vv;
      vv.q[0] = *(const uint4*)(vr + 0);
      vv.q[1] = *(const uint4*)(vr + 8);
      vv.q[2] = *(const uint4*)(vr + 16);
      vv.q[3] = *(const uint4*)(vr + 24);
#pragma unroll
      for (int d2 = 0; d2 < 32; d2++) Vt[(hi * 32 + d2) * 24 + ln] = vv.u[d2];
    }

    // scale + bias + mask + online softmax (C layout: row = r+8*hi, col = ln)
    const int jcol = j0 + ln;
#pragma unroll
    for (int r = 0; r < 8; r++) {
      const int i = i0 + r + 8 * hi;
      const int rel = i - jcol;
      float sv = s[r] * 0.125f;  // 1/sqrt(64)
      const bool valid = (rel >= 0) && (rel < R_) && (jcol >= 0);
      sv = valid ? (sv + res_pos[h * R_ + rel]) : -1e30f;
      float mx = sv;
      mx = fmaxf(mx, __shfl_xor(mx, 1, 16));
      mx = fmaxf(mx, __shfl_xor(mx, 2, 16));
      mx = fmaxf(mx, __shfl_xor(mx, 4, 16));
      mx = fmaxf(mx, __shfl_xor(mx, 8, 16));
      const float nm = fmaxf(rmax[r], mx);
      const float alpha = __expf(rmax[r] - nm);
      const float p = __expf(sv - nm);
      float ps = p;
      ps += __shfl_xor(ps, 1, 16);
      ps += __shfl_xor(ps, 2, 16);
      ps += __shfl_xor(ps, 4, 16);
      ps += __shfl_xor(ps, 8, 16);
      rsum[r] = rsum[r] * alpha + ps;
      rmax[r] = nm;
#pragma unroll
      for (int g2 = 0; g2 < 4; g2++) oacc[g2][r] *= alpha;
      Pl[(r + 8 * hi) * 16 + ln] = f2bf(p);
    }
    __syncthreads();

    // P*V: A = P (keys 0..15 valid, 16..31 zeroed), B = Vt (elements 8..15 zeroed)
    Frag16 ap, bvf[4];
    ap.q[0] = *(const uint4*)&Pl[ln * 16 + hi * 8];
    ap.q[1].x = ap.q[1].y = ap.q[1].z = ap.q[1].w = 0u;
#pragma unroll
    for (int g2 = 0; g2 < 4; g2++) {
      bvf[g2].q[0] = *(const uint4*)&Vt[(g2 * 16 + ln) * 24 + hi * 8];
      bvf[g2].q[1].x = bvf[g2].q[1].y = bvf[g2].q[1].z = bvf[g2].q[1].w = 0u;
    }
#pragma unroll
    for (int g2 = 0; g2 < 4; g2++)
      oacc[g2] = __builtin_amdgcn_wmma_f32_16x16x32_bf16(
          false, ap.v, false, bvf[g2].v, (short)0, oacc[g2], false, false);
  }

  // normalize and emit bf16 attention output [B,L,E]
#pragma unroll
  for (int g2 = 0; g2 < 4; g2++) {
#pragma unroll
    for (int r = 0; r < 8; r++) {
      const int i = i0 + r + 8 * hi;
      const float val = oacc[g2][r] / rsum[r];
      o[(size_t)(b * L_ + i) * E_ + h * DH_ + g2 * 16 + ln] = f2bf(val);
    }
  }
}

// ---------------------------------------------------------------- launch
extern "C" void kernel_launch(void* const* d_in, const int* in_sizes, int n_in,
                              void* d_out, int out_size, void* d_ws, size_t ws_size,
                              hipStream_t stream) {
  (void)in_sizes; (void)n_in; (void)out_size; (void)ws_size;
  const float* x     = (const float*)d_in[0];
  const float* res   = (const float*)d_in[1];
  const float* Wq    = (const float*)d_in[2];
  const float* bq    = (const float*)d_in[3];
  const float* Wk    = (const float*)d_in[4];
  const float* bk    = (const float*)d_in[5];
  const float* Wv    = (const float*)d_in[6];
  const float* bv    = (const float*)d_in[7];
  const float* Wo    = (const float*)d_in[8];
  const float* bo    = (const float*)d_in[9];
  const float* W1    = (const float*)d_in[10];
  const float* b1    = (const float*)d_in[11];
  const float* W2    = (const float*)d_in[12];
  const float* b2    = (const float*)d_in[13];
  const float* g_in  = (const float*)d_in[14];
  const float* b_in  = (const float*)d_in[15];
  const float* g_mlp = (const float*)d_in[16];
  const float* b_mlp = (const float*)d_in[17];

  char* ws = (char*)d_ws;
  size_t off = 0;
  auto carve = [&](size_t bytes) -> void* {
    void* p = ws + off;
    off = (off + bytes + 255) & ~(size_t)255;
    return p;
  };
  unsigned short* wq_b = (unsigned short*)carve((size_t)E_ * E_ * 2);
  unsigned short* wk_b = (unsigned short*)carve((size_t)E_ * E_ * 2);
  unsigned short* wv_b = (unsigned short*)carve((size_t)E_ * E_ * 2);
  unsigned short* wo_b = (unsigned short*)carve((size_t)E_ * E_ * 2);
  unsigned short* w1_b = (unsigned short*)carve((size_t)FF_ * E_ * 2);
  unsigned short* w2_b = (unsigned short*)carve((size_t)E_ * FF_ * 2);
  unsigned short* xn1  = (unsigned short*)carve((size_t)M_ * E_ * 2);
  unsigned short* qb   = (unsigned short*)carve((size_t)M_ * E_ * 2);
  unsigned short* kb2  = (unsigned short*)carve((size_t)M_ * E_ * 2);
  unsigned short* vb2  = (unsigned short*)carve((size_t)M_ * E_ * 2);
  unsigned short* ob   = (unsigned short*)carve((size_t)M_ * E_ * 2);
  float*          x1   = (float*)carve((size_t)M_ * E_ * 4);
  unsigned short* xn2  = (unsigned short*)carve((size_t)M_ * E_ * 2);
  unsigned short* hb   = (unsigned short*)carve((size_t)M_ * FF_ * 2);

  // weight conversions
  k_cvt<<<2048, 256, 0, stream>>>(Wq, wq_b, E_ * E_);
  k_cvt<<<2048, 256, 0, stream>>>(Wk, wk_b, E_ * E_);
  k_cvt<<<2048, 256, 0, stream>>>(Wv, wv_b, E_ * E_);
  k_cvt<<<2048, 256, 0, stream>>>(Wo, wo_b, E_ * E_);
  k_cvt<<<4096, 256, 0, stream>>>(W1, w1_b, FF_ * E_);
  k_cvt<<<4096, 256, 0, stream>>>(W2, w2_b, E_ * FF_);

  // LN1
  k_ln<<<M_, 256, 0, stream>>>(x, g_in, b_in, xn1);

  const dim3 gEE(M_ / 128, E_ / 128);
  const dim3 gEF(M_ / 128, FF_ / 128);

  // Q, K, V projections (bf16 out)
  k_gemm<false, false, false, true><<<gEE, 256, 0, stream>>>(xn1, wq_b, bq, nullptr, nullptr, qb, M_, E_, E_);
  k_gemm<false, false, false, true><<<gEE, 256, 0, stream>>>(xn1, wk_b, bk, nullptr, nullptr, kb2, M_, E_, E_);
  k_gemm<false, false, false, true><<<gEE, 256, 0, stream>>>(xn1, wv_b, bv, nullptr, nullptr, vb2, M_, E_, E_);

  // banded attention
  k_attn<<<B_ * H_ * (L_ / 16), 32, 0, stream>>>(qb, kb2, vb2, res, ob);

  // x1 = x + o @ Wo^T + bo  (f32 out)
  k_gemm<false, true, true, false><<<gEE, 256, 0, stream>>>(ob, wo_b, bo, x, x1, nullptr, M_, E_, E_);

  // LN2
  k_ln<<<M_, 256, 0, stream>>>(x1, g_mlp, b_mlp, xn2);

  // h = relu(xn2 @ W1^T + b1)  (bf16 out)
  k_gemm<true, false, false, true><<<gEF, 256, 0, stream>>>(xn2, w1_b, b1, nullptr, nullptr, hb, M_, FF_, E_);

  // out = x1 + h @ W2^T + b2  (f32 out)
  k_gemm<false, true, true, false><<<gEE, 256, 0, stream>>>(hb, w2_b, b2, x1, (float*)d_out, nullptr, M_, E_, FF_);
}
